// MultiHeadDiffAttention_79491254714581
// MI455X (gfx1250) — matrix operand order repaired and checked by reading, not compiled, not run on hardware
//
#include <hip/hip_runtime.h>
#include <hip/hip_bf16.h>

#define T_SEQ   2048
#define C_DIM   1024
#define H_NUM   16
#define HS      64
#define DV      128           // 2*HS
#define BATCH   2
#define BT      4096          // BATCH*T_SEQ
#define LAMBDA_INIT 0.7778700995592556f

typedef _Float16 h8  __attribute__((ext_vector_type(8)));
typedef _Float16 h16 __attribute__((ext_vector_type(16)));
typedef float    f8  __attribute__((ext_vector_type(8)));

// ---------------------------------------------------------------- helpers ---

__device__ __forceinline__ h16 combine16(h8 lo, h8 hi) {
    return __builtin_shufflevector(lo, hi, 0,1,2,3,4,5,6,7,8,9,10,11,12,13,14,15);
}

// B fragment (32x16, f16): lane n = lane&15 selects column; lanes 0-15 hold
// K=0..15, lanes 16-31 hold K=16..31; per-lane 16 contraction values are
// contiguous in memory when the matrix is stored [N][K] row-major.
__device__ __forceinline__ h16 load_b16(const _Float16* p) {
    h8 lo = *(const h8*)p;
    h8 hi = *(const h8*)(p + 8);
    return combine16(lo, hi);
}

// A fragment (16x32, f16, row-major source, rowk = &A[m][kb]):
// lanes 0-15 (lh=0): K = 0..15; lanes 16-31 (lh=1): K = 16..31
__device__ __forceinline__ h16 load_afrag(const _Float16* rowk, int lh) {
    h8 lo = *(const h8*)(rowk + lh * 16);
    h8 hi = *(const h8*)(rowk + lh * 16 + 8);
    return combine16(lo, hi);
}

__device__ __forceinline__ f8 wmma_f16(h16 a, h16 b, f8 c) {
    return __builtin_amdgcn_wmma_f32_16x16x32_f16(false, a, false, b,
                                                  (short)0, c, false, false);
}

// ---------------------------------------------------------------- convert ---

__global__ void cvt_f32_f16(const float* __restrict__ in,
                            _Float16* __restrict__ out, int n) {
    int i = blockIdx.x * blockDim.x + threadIdx.x;
    if (i < n) out[i] = (_Float16)in[i];
}

// ------------------------------------------------------------------- GEMM ---
// OUT[M,N] = A[M,K] @ W[N,K]^T, one wave per 16x128 output tile.

template <bool OUT16>
__global__ __launch_bounds__(32, 1)
void gemm_wmma_kernel(const _Float16* __restrict__ A,
                      const _Float16* __restrict__ W,
                      void* __restrict__ outp,
                      int M, int N, int K, int mtiles) {
    const int lane = threadIdx.x & 31;
    const int lh   = lane >> 4;   // which 16-lane half
    const int ln   = lane & 15;   // A-row / B-col / D-col within tile
    const int blk  = blockIdx.x;
    const int mt   = blk % mtiles;
    const int nt   = blk / mtiles;

    const _Float16* arow = A + (size_t)(mt * 16 + ln) * K;
    f8 acc[8] = {};

    for (int kb = 0; kb < K; kb += 32) {
        __builtin_prefetch(arow + kb + 128, 0, 1);
        h16 a = load_afrag(arow + kb, lh);
#pragma unroll
        for (int s = 0; s < 8; ++s) {
            const int n = nt * 128 + s * 16 + ln;
            h16 bf = load_b16(W + (size_t)n * K + kb + lh * 16);
            acc[s] = wmma_f16(a, bf, acc[s]);
        }
    }

#pragma unroll
    for (int s = 0; s < 8; ++s) {
#pragma unroll
        for (int r = 0; r < 8; ++r) {
            const int row = mt * 16 + r + 8 * lh;   // C/D layout: M = r + 8*half
            const int col = nt * 128 + s * 16 + ln;
            const float v = acc[s][r];
            if (OUT16) ((_Float16*)outp)[(size_t)row * N + col] = (_Float16)v;
            else       ((float*)outp)[(size_t)row * N + col] = v;
        }
    }
}

// -------------------------------------------------------- flash attention ---
// One wave per (b, h, 16-query tile). Causal softmax(Q K^T / 8) @ V -> Y (f32).

__global__ __launch_bounds__(32, 1)
void diff_attn_pass(const _Float16* __restrict__ Q,
                    const _Float16* __restrict__ Km,
                    const _Float16* __restrict__ V,
                    float* __restrict__ Y) {
    const int lane = threadIdx.x & 31;
    const int lh   = lane >> 4;
    const int ln   = lane & 15;
    const int qt   = blockIdx.x;
    const int h    = blockIdx.y;
    const int b    = blockIdx.z;
    const int qbase = qt * 16;

    __shared__ _Float16 VT[128 * 40];   // V tile transposed [d][key], pad 40
    __shared__ _Float16 P[16 * 32];     // probabilities tile [q][key]

    const _Float16* qrow = Q + ((size_t)(b * T_SEQ + qbase + ln)) * C_DIM + h * HS;
    h16 qa0 = load_afrag(qrow, lh);        // K dims 0..31
    h16 qa1 = load_afrag(qrow + 32, lh);   // K dims 32..63

    f8 acc[8] = {};
    float m[8], l[8];
#pragma unroll
    for (int r = 0; r < 8; ++r) { m[r] = -1e30f; l[r] = 0.f; }

    const float scale = 0.125f;               // 1/sqrt(64)
    const int nkb = (qbase + 47) >> 5;        // 32-key blocks covering [0, qbase+15]

    for (int kb = 0; kb < nkb; ++kb) {
        const int kstart = kb * 32;

        // ---- stage V tile (32 keys x 128 dims) transposed into LDS --------
        {
            const _Float16* vrow =
                V + ((size_t)(b * T_SEQ + kstart + lane)) * (2 * C_DIM) + h * DV;
#pragma unroll
            for (int c = 0; c < 16; ++c) {
                h8 g = *(const h8*)(vrow + c * 8);
#pragma unroll
                for (int e = 0; e < 8; ++e) VT[(c * 8 + e) * 40 + lane] = g[e];
            }
        }
        __syncthreads();

        // ---- scores: two 16x16 tiles, contraction 64 = 2 WMMA each --------
        f8 s0 = {}, s1 = {};
        {
            const int key0 = kstart + ln;
            const _Float16* kr0 = Km + ((size_t)(b * T_SEQ + key0)) * C_DIM + h * HS;
            s0 = wmma_f16(qa0, load_b16(kr0 + lh * 16), s0);
            s0 = wmma_f16(qa1, load_b16(kr0 + 32 + lh * 16), s0);
            const int key1 = kstart + 16 + ln;
            const _Float16* kr1 = Km + ((size_t)(b * T_SEQ + key1)) * C_DIM + h * HS;
            s1 = wmma_f16(qa0, load_b16(kr1 + lh * 16), s1);
            s1 = wmma_f16(qa1, load_b16(kr1 + 32 + lh * 16), s1);
        }

        // ---- causal mask + online softmax (row = r + 8*lh) ----------------
        float resc[8];
#pragma unroll
        for (int r = 0; r < 8; ++r) {
            const int row = qbase + r + 8 * lh;
            float a0 = s0[r] * scale;
            float a1 = s1[r] * scale;
            if (kstart + ln > row)      a0 = -1e30f;
            if (kstart + 16 + ln > row) a1 = -1e30f;
            float t = fmaxf(a0, a1);
            t = fmaxf(t, __shfl_xor(t, 1, 32));
            t = fmaxf(t, __shfl_xor(t, 2, 32));
            t = fmaxf(t, __shfl_xor(t, 4, 32));
            t = fmaxf(t, __shfl_xor(t, 8, 32));
            const float mn = fmaxf(m[r], t);
            const float p0 = __expf(a0 - mn);
            const float p1 = __expf(a1 - mn);
            float rs = p0 + p1;
            rs += __shfl_xor(rs, 1, 32);
            rs += __shfl_xor(rs, 2, 32);
            rs += __shfl_xor(rs, 4, 32);
            rs += __shfl_xor(rs, 8, 32);
            resc[r] = __expf(m[r] - mn);
            l[r] = l[r] * resc[r] + rs;
            m[r] = mn;
            const int prow = r + 8 * lh;
            P[prow * 32 + ln]      = (_Float16)p0;
            P[prow * 32 + 16 + ln] = (_Float16)p1;
        }
#pragma unroll
        for (int c = 0; c < 8; ++c) {
#pragma unroll
            for (int r = 0; r < 8; ++r) acc[c][r] *= resc[r];
        }
        __syncthreads();

        // ---- P @ V : A = P (16x32 from LDS), B = VT columns ---------------
        h16 pa = load_afrag(&P[ln * 32], lh);
#pragma unroll
        for (int c = 0; c < 8; ++c) {
            h16 vb = load_b16(&VT[(c * 16 + ln) * 40 + lh * 16]);
            acc[c] = wmma_f16(pa, vb, acc[c]);
        }
        __syncthreads();
    }

    // ---- normalize and store ---------------------------------------------
#pragma unroll
    for (int c = 0; c < 8; ++c) {
#pragma unroll
        for (int r = 0; r < 8; ++r) {
            const int row = qbase + r + 8 * lh;
            const int d   = c * 16 + ln;
            Y[((size_t)((b * H_NUM + h) * T_SEQ) + row) * DV + d] = acc[c][r] / l[r];
        }
    }
}

// -------------------------------------------- groupnorm + lambda combine ---
// One block per (b,h). y = y1 - lam*y2; GroupNorm over (T, 4 ch/group);
// affine * gamma * (1 - lambda_init); write f16 [B,T,2C] layout.

__global__ void groupnorm_combine(const float* __restrict__ Y1,
                                  const float* __restrict__ Y2,
                                  const float* __restrict__ lq1,
                                  const float* __restrict__ lk1,
                                  const float* __restrict__ lq2,
                                  const float* __restrict__ lk2,
                                  const float* __restrict__ gnw,
                                  const float* __restrict__ gnb,
                                  const float* __restrict__ gam,
                                  _Float16* __restrict__ Yout) {
    const int bh = blockIdx.x;
    const int b  = bh >> 4;
    const int h  = bh & 15;
    const int tid = threadIdx.x;

    const float x1 = lq1[h] * lk1[h];
    const float x2 = lq2[h] * lk2[h];
    const float lam = 1.f / (1.f + __expf(-x1)) - 1.f / (1.f + __expf(-x2))
                      + LAMBDA_INIT;

    const float* y1 = Y1 + (size_t)bh * T_SEQ * DV;
    const float* y2 = Y2 + (size_t)bh * T_SEQ * DV;

    float s = 0.f, ss = 0.f;
    for (int i = tid; i < T_SEQ * DV; i += 256) {   // i&127 == tid&127 always
        const float v = y1[i] - lam * y2[i];
        s += v; ss += v * v;
    }
    __shared__ float S[256], SQ[256];
    __shared__ float MU[32], RS[32];
    S[tid] = s; SQ[tid] = ss;
    __syncthreads();
    if (tid < 32) {                                  // group g = tid, d in [4g,4g+4)
        float a = 0.f, q = 0.f;
#pragma unroll
        for (int j = 0; j < 4; ++j) {
            a += S[tid * 4 + j] + S[128 + tid * 4 + j];
            q += SQ[tid * 4 + j] + SQ[128 + tid * 4 + j];
        }
        const float inv = 1.f / (float)(T_SEQ * 4);
        const float mu  = a * inv;
        const float var = q * inv - mu * mu;
        MU[tid] = mu;
        RS[tid] = rsqrtf(var + 1e-5f);
    }
    __syncthreads();

    const float post = 1.f - LAMBDA_INIT;
    for (int i = tid; i < T_SEQ * DV; i += 256) {
        const int d = i & 127;
        const int g = d >> 2;
        const int t = i >> 7;
        const float v  = y1[i] - lam * y2[i];
        float vn = (v - MU[g]) * RS[g];
        vn = (vn * gnw[d] + gnb[d]) * gam[d] * post;
        Yout[((size_t)(b * T_SEQ + t)) * (2 * C_DIM) + h * DV + d] = (_Float16)vn;
    }
}

// ---------------------------------------------------------------- launch ---

extern "C" void kernel_launch(void* const* d_in, const int* in_sizes, int n_in,
                              void* d_out, int out_size, void* d_ws, size_t ws_size,
                              hipStream_t stream) {
    (void)in_sizes; (void)n_in; (void)out_size; (void)ws_size;

    const float* x   = (const float*)d_in[0];
    const float* Wq1 = (const float*)d_in[1];
    const float* Wq2 = (const float*)d_in[2];
    const float* Wk1 = (const float*)d_in[3];
    const float* Wk2 = (const float*)d_in[4];
    const float* Wv  = (const float*)d_in[5];
    const float* Wc  = (const float*)d_in[6];
    const float* gnw = (const float*)d_in[7];
    const float* gnb = (const float*)d_in[8];
    const float* gam = (const float*)d_in[9];
    const float* lq1 = (const float*)d_in[10];
    const float* lk1 = (const float*)d_in[11];
    const float* lq2 = (const float*)d_in[12];
    const float* lk2 = (const float*)d_in[13];
    float* out = (float*)d_out;

    size_t off = 0;
    auto take = [&](size_t bytes) -> char* {
        char* p = (char*)d_ws + off; off += bytes; return p;
    };
    _Float16* xh   = (_Float16*)take((size_t)BT * C_DIM * 2);
    _Float16* wq1h = (_Float16*)take((size_t)C_DIM * C_DIM * 2);
    _Float16* wq2h = (_Float16*)take((size_t)C_DIM * C_DIM * 2);
    _Float16* wk1h = (_Float16*)take((size_t)C_DIM * C_DIM * 2);
    _Float16* wk2h = (_Float16*)take((size_t)C_DIM * C_DIM * 2);
    _Float16* wvh  = (_Float16*)take((size_t)2 * C_DIM * C_DIM * 2);
    _Float16* wch  = (_Float16*)take((size_t)2 * C_DIM * C_DIM * 2);
    _Float16* q1h  = (_Float16*)take((size_t)BT * C_DIM * 2);
    _Float16* q2h  = (_Float16*)take((size_t)BT * C_DIM * 2);
    _Float16* k1h  = (_Float16*)take((size_t)BT * C_DIM * 2);
    _Float16* k2h  = (_Float16*)take((size_t)BT * C_DIM * 2);
    _Float16* vh   = (_Float16*)take((size_t)BT * 2 * C_DIM * 2);
    float*    y1   = (float*)take((size_t)BATCH * H_NUM * T_SEQ * DV * 4);
    float*    y2   = (float*)take((size_t)BATCH * H_NUM * T_SEQ * DV * 4);
    // q1h/q2h are dead after the attention passes: alias the 16MB f16
    // normalized-activation buffer over them.
    _Float16* y2h = q1h;

    // ---- f32 -> f16 conversions ------------------------------------------
    auto cvt = [&](const float* src, _Float16* dst, int n) {
        cvt_f32_f16<<<(n + 255) / 256, 256, 0, stream>>>(src, dst, n);
    };
    cvt(x,   xh,   BT * C_DIM);
    cvt(Wq1, wq1h, C_DIM * C_DIM);
    cvt(Wq2, wq2h, C_DIM * C_DIM);
    cvt(Wk1, wk1h, C_DIM * C_DIM);
    cvt(Wk2, wk2h, C_DIM * C_DIM);
    cvt(Wv,  wvh,  2 * C_DIM * C_DIM);
    cvt(Wc,  wch,  2 * C_DIM * C_DIM);

    // ---- projections (16x128 tiles per wave) -----------------------------
    const int MT = BT / 16;   // 256 M-tiles
    gemm_wmma_kernel<true><<<MT * (C_DIM / 128), 32, 0, stream>>>(
        xh, wq1h, q1h, BT, C_DIM, C_DIM, MT);
    gemm_wmma_kernel<true><<<MT * (C_DIM / 128), 32, 0, stream>>>(
        xh, wq2h, q2h, BT, C_DIM, C_DIM, MT);
    gemm_wmma_kernel<true><<<MT * (C_DIM / 128), 32, 0, stream>>>(
        xh, wk1h, k1h, BT, C_DIM, C_DIM, MT);
    gemm_wmma_kernel<true><<<MT * (C_DIM / 128), 32, 0, stream>>>(
        xh, wk2h, k2h, BT, C_DIM, C_DIM, MT);
    gemm_wmma_kernel<true><<<MT * (2 * C_DIM / 128), 32, 0, stream>>>(
        xh, wvh, vh, BT, 2 * C_DIM, C_DIM, MT);

    // ---- two flash-attention passes --------------------------------------
    dim3 agrid(T_SEQ / 16, H_NUM, BATCH);
    diff_attn_pass<<<agrid, 32, 0, stream>>>(q1h, k1h, vh, y1);
    diff_attn_pass<<<agrid, 32, 0, stream>>>(q2h, k2h, vh, y2);

    // ---- lambda combine + groupnorm + affine -----------------------------
    groupnorm_combine<<<BATCH * H_NUM, 256, 0, stream>>>(
        y1, y2, lq1, lk1, lq2, lk2, gnw, gnb, gam, y2h);

    // ---- output projection (f32 out) -------------------------------------
    gemm_wmma_kernel<false><<<MT * (C_DIM / 128), 32, 0, stream>>>(
        y2h, wch, out, BT, C_DIM, 2 * C_DIM, MT);
}
